// AttentionBlock_10385230922236
// MI455X (gfx1250) — matrix operand order repaired and checked by reading, not compiled, not run on hardware
//
#include <hip/hip_runtime.h>

// ---------------------------------------------------------------------------
// AttentionBlock for MI455X (gfx1250): all GEMMs via v_wmma_f32_16x16x32_bf16.
// FLOPs ~172G, mandatory HBM ~268MB -> bf16 intermediates keep us near the
// memory roofline while WMMA eats the math.
// ---------------------------------------------------------------------------

typedef __bf16 bf16;
typedef __bf16 v16bf __attribute__((ext_vector_type(16)));
typedef __bf16 v8bf  __attribute__((ext_vector_type(8)));
typedef float  v8f   __attribute__((ext_vector_type(8)));

#define DEVINL __device__ __forceinline__

static constexpr int B_ = 256;
static constexpr int C_ = 512;
static constexpr int S_ = 256;
static constexpr float GN_EPS = 1e-5f;
static constexpr float ATT_SCALE = 0.04419417382415922f; // 1/sqrt(512)

DEVINL v8f zero8() {
  v8f z;
#pragma unroll
  for (int i = 0; i < 8; ++i) z[i] = 0.f;
  return z;
}

// A fragment: 16x32 bf16, row-major source, lda elements.
// lane<16 : row m=lane,    K = k0+0..7  and k0+16..23
// lane>=16: row m=lane-16, K = k0+8..15 and k0+24..31
DEVINL v16bf load_a(const bf16* __restrict__ A, int lda, int m0, int k0, int lane) {
  const int m  = m0 + (lane & 15);
  const int kb = k0 + ((lane >> 4) << 3);
  const bf16* p = A + (size_t)m * lda + kb;
  union { v16bf v; v8bf h[2]; } u;
  u.h[0] = *reinterpret_cast<const v8bf*>(p);
  u.h[1] = *reinterpret_cast<const v8bf*>(p + 16);
  return u.v;
}

// B fragment: 32x16 bf16 loaded from B-transposed storage Bt[N][ldb] (=K dim).
// lane<16 : col n=lane,    K = k0+0..7 and k0+16..23 ; lane>=16: K = k0+8..15 and k0+24..31
DEVINL v16bf load_b(const bf16* __restrict__ Bt, int ldb, int n0, int k0, int lane) {
  const int n  = n0 + (lane & 15);
  const int kb = k0 + ((lane >> 4) << 3);
  const bf16* p = Bt + (size_t)n * ldb + kb;
  union { v16bf v; v8bf h[2]; } u;
  u.h[0] = *reinterpret_cast<const v8bf*>(p);
  u.h[1] = *reinterpret_cast<const v8bf*>(p + 16);
  return u.v;
}

DEVINL v8f wmma_bf16(v16bf a, v16bf b, v8f c) {
  return __builtin_amdgcn_wmma_f32_16x16x32_bf16(false, a, false, b, (short)0, c,
                                                 false, false);
}

// D tile element i of lane l = D[tm + i + 8*(l>>4)][tn + (l&15)]

// Store 16x16 f32 acc tile transposed (dst[n][m]) as bf16, +bias[m].
DEVINL void store_T(bf16* __restrict__ dst, int ldd, int tm, int tn, v8f acc,
                    const float* __restrict__ bias, int lane) {
  const int n  = tn + (lane & 15);
  const int mb = tm + ((lane >> 4) << 3);
  v8bf o;
#pragma unroll
  for (int i = 0; i < 8; ++i) o[i] = (bf16)(acc[i] + bias[mb + i]);
  *reinterpret_cast<v8bf*>(dst + (size_t)n * ldd + mb) = o;
}

DEVINL void store_T_nb(bf16* __restrict__ dst, int ldd, int tm, int tn, v8f acc,
                       int lane) {
  const int n  = tn + (lane & 15);
  const int mb = tm + ((lane >> 4) << 3);
  v8bf o;
#pragma unroll
  for (int i = 0; i < 8; ++i) o[i] = (bf16)acc[i];
  *reinterpret_cast<v8bf*>(dst + (size_t)n * ldd + mb) = o;
}

// Store tile in natural [m][n] order (bf16), +bias[m].
DEVINL void store_N(bf16* __restrict__ dst, int ldn, int tm, int tn, v8f acc,
                    const float* __restrict__ bias, int lane) {
  const int n  = tn + (lane & 15);
  const int mb = tm + ((lane >> 4) << 3);
#pragma unroll
  for (int i = 0; i < 8; ++i)
    dst[(size_t)(mb + i) * ldn + n] = (bf16)(acc[i] + bias[mb + i]);
}

// Final f32 store: out = acc + bias[m] + x[m][n]
DEVINL void store_res(float* __restrict__ ob, const float* __restrict__ xb,
                      const float* __restrict__ bp, int tm, int tn, v8f acc,
                      int lane) {
  const int n  = tn + (lane & 15);
  const int mb = tm + ((lane >> 4) << 3);
#pragma unroll
  for (int i = 0; i < 8; ++i) {
    const int m = mb + i;
    ob[(size_t)m * S_ + n] = acc[i] + bp[m] + xb[(size_t)m * S_ + n];
  }
}

// ---------------------------------------------------------------------------
// Kernel 1: convert the four 512x512 fp32 weight matrices to bf16 scratch.
__global__ void k_cvt_w(const float* __restrict__ wq, const float* __restrict__ wk,
                        const float* __restrict__ wv, const float* __restrict__ wp,
                        bf16* __restrict__ dst) {
  const int idx = blockIdx.x * 256 + threadIdx.x;   // 0 .. 4*262144-1
  const float* src = (idx < 262144) ? wq
                   : (idx < 524288) ? wk
                   : (idx < 786432) ? wv : wp;
  dst[idx] = (bf16)src[idx & 262143];
}

// ---------------------------------------------------------------------------
// Kernel 2: GroupNorm(32,512) -> h stored transposed [b][s][c] in bf16.
// One block per (b,g); thread t owns pixel s=t across the 16 group channels.
__global__ void k_gn(const float* __restrict__ x, const float* __restrict__ gw,
                     const float* __restrict__ gb, bf16* __restrict__ h_t) {
  const int b = blockIdx.x >> 5;
  const int g = blockIdx.x & 31;
  const int t = threadIdx.x;           // s index
  const float* xp = x + ((size_t)b * C_ + g * 16) * S_;
  float vals[16];
  float sum = 0.f, sq = 0.f;
#pragma unroll
  for (int i = 0; i < 16; ++i) {
    const float v = xp[(size_t)i * S_ + t];
    vals[i] = v;
    sum += v;
    sq += v * v;
  }
  __shared__ float s_sum[256], s_sq[256];
  s_sum[t] = sum; s_sq[t] = sq;
  __syncthreads();
  for (int off = 128; off > 0; off >>= 1) {
    if (t < off) { s_sum[t] += s_sum[t + off]; s_sq[t] += s_sq[t + off]; }
    __syncthreads();
  }
  const float inv_n = 1.f / 4096.f;
  const float mean  = s_sum[0] * inv_n;
  const float var   = s_sq[0] * inv_n - mean * mean;
  const float rstd  = rsqrtf(var + GN_EPS);
  v8bf o0, o1;
#pragma unroll
  for (int i = 0; i < 8; ++i) {
    o0[i] = (bf16)((vals[i] - mean) * rstd * gw[g * 16 + i] + gb[g * 16 + i]);
    o1[i] = (bf16)((vals[i + 8] - mean) * rstd * gw[g * 16 + 8 + i] + gb[g * 16 + 8 + i]);
  }
  bf16* hp = h_t + ((size_t)b * S_ + t) * C_ + g * 16;
  *reinterpret_cast<v8bf*>(hp)     = o0;
  *reinterpret_cast<v8bf*>(hp + 8) = o1;
}

// ---------------------------------------------------------------------------
// Kernel 3: q/k/v = W*h + b. grid = (16 tiles, B, 3{q,k,v}); block=8 waves,
// each wave a 32x32 register tile (2x2 WMMA accumulators), K=512.
// q,k stored transposed [b][s][c]; v stored natural [b][c][s].
__global__ void k_qkv(const bf16* __restrict__ wq_b, const bf16* __restrict__ wk_b,
                      const bf16* __restrict__ wv_b,
                      const float* __restrict__ bq, const float* __restrict__ bk,
                      const float* __restrict__ bv,
                      const bf16* __restrict__ h_t,
                      bf16* __restrict__ q_t, bf16* __restrict__ k_t,
                      bf16* __restrict__ v) {
  const int which = blockIdx.z;
  const bf16*  A    = (which == 0) ? wq_b : (which == 1) ? wk_b : wv_b;
  const float* bias = (which == 0) ? bq   : (which == 1) ? bk   : bv;
  const int b    = blockIdx.y;
  const int bm   = blockIdx.x >> 2, bn = blockIdx.x & 3;
  const int wave = threadIdx.x >> 5, lane = threadIdx.x & 31;
  const int m0 = bm * 128 + (wave >> 1) * 32;
  const int n0 = bn * 64 + (wave & 1) * 32;
  const bf16* Bt = h_t + (size_t)b * S_ * C_;   // [s][c]
  v8f c00 = zero8(), c01 = zero8(), c10 = zero8(), c11 = zero8();
  for (int k = 0; k < 512; k += 32) {
    const v16bf a0 = load_a(A, 512, m0, k, lane);
    const v16bf a1 = load_a(A, 512, m0 + 16, k, lane);
    const v16bf b0 = load_b(Bt, 512, n0, k, lane);
    const v16bf b1 = load_b(Bt, 512, n0 + 16, k, lane);
    c00 = wmma_bf16(a0, b0, c00);
    c10 = wmma_bf16(a1, b0, c10);
    c01 = wmma_bf16(a0, b1, c01);
    c11 = wmma_bf16(a1, b1, c11);
  }
  if (which < 2) {
    bf16* dst = ((which == 0) ? q_t : k_t) + (size_t)b * S_ * C_;
    store_T(dst, 512, m0, n0, c00, bias, lane);
    store_T(dst, 512, m0 + 16, n0, c10, bias, lane);
    store_T(dst, 512, m0, n0 + 16, c01, bias, lane);
    store_T(dst, 512, m0 + 16, n0 + 16, c11, bias, lane);
  } else {
    bf16* dst = v + (size_t)b * C_ * S_;
    store_N(dst, 256, m0, n0, c00, bias, lane);
    store_N(dst, 256, m0 + 16, n0, c10, bias, lane);
    store_N(dst, 256, m0, n0 + 16, c01, bias, lane);
    store_N(dst, 256, m0 + 16, n0 + 16, c11, bias, lane);
  }
}

// ---------------------------------------------------------------------------
// Kernel 4: scores L[s][t] = qT k / sqrt(C), row softmax over t, write
// P transposed [b][t][s] in bf16 (= B^T layout for the AV GEMM).
// Block = one batch, 16 score rows, all 256 cols. Wave w owns cols w*32..+31.
__global__ void k_att(const bf16* __restrict__ q_t, const bf16* __restrict__ k_t,
                      bf16* __restrict__ attT) {
  const int b    = blockIdx.y;
  const int m0   = blockIdx.x * 16;    // s row base
  const int wave = threadIdx.x >> 5, lane = threadIdx.x & 31;
  const int tid  = threadIdx.x;
  const bf16* Aq = q_t + (size_t)b * S_ * C_;
  const bf16* Bk = k_t + (size_t)b * S_ * C_;
  __shared__ float Lsh[16 * 256];
  __shared__ float red[16 * 16];

  const int n0 = wave * 32;
  v8f c0 = zero8(), c1 = zero8();
  for (int k = 0; k < 512; k += 32) {
    const v16bf a  = load_a(Aq, 512, m0, k, lane);
    const v16bf b0 = load_b(Bk, 512, n0, k, lane);
    const v16bf b1 = load_b(Bk, 512, n0 + 16, k, lane);
    c0 = wmma_bf16(a, b0, c0);
    c1 = wmma_bf16(a, b1, c1);
  }
  {
    const int n  = n0 + (lane & 15);
    const int rb = (lane >> 4) << 3;
#pragma unroll
    for (int i = 0; i < 8; ++i) {
      Lsh[(rb + i) * 256 + n]      = c0[i] * ATT_SCALE;
      Lsh[(rb + i) * 256 + n + 16] = c1[i] * ATT_SCALE;
    }
  }
  __syncthreads();

  const int r = tid >> 4, sg = tid & 15;   // 16 threads per row
  float mx = -3.4e38f;
#pragma unroll
  for (int j = 0; j < 16; ++j) mx = fmaxf(mx, Lsh[r * 256 + sg * 16 + j]);
  red[r * 16 + sg] = mx;
  __syncthreads();
  float rowmax = red[r * 16];
#pragma unroll
  for (int j = 1; j < 16; ++j) rowmax = fmaxf(rowmax, red[r * 16 + j]);
  __syncthreads();
  float ps = 0.f;
#pragma unroll
  for (int j = 0; j < 16; ++j) {
    const float e = __expf(Lsh[r * 256 + sg * 16 + j] - rowmax);
    Lsh[r * 256 + sg * 16 + j] = e;
    ps += e;
  }
  red[r * 16 + sg] = ps;
  __syncthreads();
  float rs = 0.f;
#pragma unroll
  for (int j = 0; j < 16; ++j) rs += red[r * 16 + j];
  const float inv = 1.f / rs;
  bf16* ap = attT + (size_t)b * S_ * S_;
#pragma unroll
  for (int j = 0; j < 16; ++j) {
    const int t_ = sg * 16 + j;
    ap[(size_t)t_ * 256 + (m0 + r)] = (bf16)(Lsh[r * 256 + t_] * inv);
  }
}

// ---------------------------------------------------------------------------
// Kernel 5: o[c][t] = sum_s v[c][s] * P[t][s];  A = v [c][s], Bt = attT [t][s].
// Output stored transposed [b][t][c] (reuses the h buffer).
__global__ void k_out(const bf16* __restrict__ v, const bf16* __restrict__ attT,
                      bf16* __restrict__ o_t) {
  const int b    = blockIdx.y;
  const int bm   = blockIdx.x >> 2, bn = blockIdx.x & 3;
  const int wave = threadIdx.x >> 5, lane = threadIdx.x & 31;
  const int m0 = bm * 128 + (wave >> 1) * 32;
  const int n0 = bn * 64 + (wave & 1) * 32;
  const bf16* A  = v    + (size_t)b * C_ * S_;
  const bf16* Bt = attT + (size_t)b * S_ * S_;
  v8f c00 = zero8(), c01 = zero8(), c10 = zero8(), c11 = zero8();
  for (int k = 0; k < 256; k += 32) {
    const v16bf a0 = load_a(A, 256, m0, k, lane);
    const v16bf a1 = load_a(A, 256, m0 + 16, k, lane);
    const v16bf b0 = load_b(Bt, 256, n0, k, lane);
    const v16bf b1 = load_b(Bt, 256, n0 + 16, k, lane);
    c00 = wmma_bf16(a0, b0, c00);
    c10 = wmma_bf16(a1, b0, c10);
    c01 = wmma_bf16(a0, b1, c01);
    c11 = wmma_bf16(a1, b1, c11);
  }
  bf16* dst = o_t + (size_t)b * S_ * C_;   // [t][c]
  store_T_nb(dst, 512, m0, n0, c00, lane);
  store_T_nb(dst, 512, m0 + 16, n0, c10, lane);
  store_T_nb(dst, 512, m0, n0 + 16, c01, lane);
  store_T_nb(dst, 512, m0 + 16, n0 + 16, c11, lane);
}

// ---------------------------------------------------------------------------
// Kernel 6: out = wp * o + bp + x  (fp32 output).
__global__ void k_proj(const bf16* __restrict__ wp_b, const float* __restrict__ bp,
                       const bf16* __restrict__ o_t, const float* __restrict__ x,
                       float* __restrict__ out) {
  const int b    = blockIdx.y;
  const int bm   = blockIdx.x >> 2, bn = blockIdx.x & 3;
  const int wave = threadIdx.x >> 5, lane = threadIdx.x & 31;
  const int m0 = bm * 128 + (wave >> 1) * 32;
  const int n0 = bn * 64 + (wave & 1) * 32;
  const bf16* Bt = o_t + (size_t)b * S_ * C_;   // [t][c]
  v8f c00 = zero8(), c01 = zero8(), c10 = zero8(), c11 = zero8();
  for (int k = 0; k < 512; k += 32) {
    const v16bf a0 = load_a(wp_b, 512, m0, k, lane);
    const v16bf a1 = load_a(wp_b, 512, m0 + 16, k, lane);
    const v16bf b0 = load_b(Bt, 512, n0, k, lane);
    const v16bf b1 = load_b(Bt, 512, n0 + 16, k, lane);
    c00 = wmma_bf16(a0, b0, c00);
    c10 = wmma_bf16(a1, b0, c10);
    c01 = wmma_bf16(a0, b1, c01);
    c11 = wmma_bf16(a1, b1, c11);
  }
  const float* xb = x + (size_t)b * C_ * S_;
  float* ob = out + (size_t)b * C_ * S_;
  store_res(ob, xb, bp, m0, n0, c00, lane);
  store_res(ob, xb, bp, m0 + 16, n0, c10, lane);
  store_res(ob, xb, bp, m0, n0 + 16, c01, lane);
  store_res(ob, xb, bp, m0 + 16, n0 + 16, c11, lane);
}

// ---------------------------------------------------------------------------
extern "C" void kernel_launch(void* const* d_in, const int* in_sizes, int n_in,
                              void* d_out, int out_size, void* d_ws, size_t ws_size,
                              hipStream_t stream) {
  (void)in_sizes; (void)n_in; (void)out_size; (void)ws_size;
  const float* x   = (const float*)d_in[0];
  const float* gnw = (const float*)d_in[1];
  const float* gnb = (const float*)d_in[2];
  const float* wq  = (const float*)d_in[3];
  const float* bq  = (const float*)d_in[4];
  const float* wk  = (const float*)d_in[5];
  const float* bk  = (const float*)d_in[6];
  const float* wv  = (const float*)d_in[7];
  const float* bv  = (const float*)d_in[8];
  const float* wp  = (const float*)d_in[9];
  const float* bp  = (const float*)d_in[10];
  float* out = (float*)d_out;

  // bf16 scratch layout
  const size_t WM  = (size_t)512 * 512;            // one weight matrix
  const size_t BSC = (size_t)B_ * S_ * C_;         // one activation tensor
  bf16* ws   = (bf16*)d_ws;
  bf16* wq_b = ws;
  bf16* wk_b = wq_b + WM;
  bf16* wv_b = wk_b + WM;
  bf16* wp_b = wv_b + WM;
  bf16* h_t  = wp_b + WM;        // [b][s][c]; reused later as o_t [b][t][c]
  bf16* q_t  = h_t + BSC;        // [b][s][c]
  bf16* k_t  = q_t + BSC;        // [b][t][c]
  bf16* v    = k_t + BSC;        // [b][c][s]
  bf16* attT = v + BSC;          // [b][t][s]

  k_cvt_w<<<4096, 256, 0, stream>>>(wq, wk, wv, wp, ws);
  k_gn<<<B_ * 32, 256, 0, stream>>>(x, gnw, gnb, h_t);
  k_qkv<<<dim3(16, B_, 3), 256, 0, stream>>>(wq_b, wk_b, wv_b, bq, bk, bv, h_t,
                                             q_t, k_t, v);
  k_att<<<dim3(16, B_), 256, 0, stream>>>(q_t, k_t, attT);
  k_out<<<dim3(16, B_), 256, 0, stream>>>(v, attT, h_t);   // h_t reused as o_t
  k_proj<<<dim3(16, B_), 256, 0, stream>>>(wp_b, bp, h_t, x, out);
}
